// MyLSTM_36962488549934
// MI455X (gfx1250) — matrix operand order repaired and checked by reading, not compiled, not run on hardware
//
#include <hip/hip_runtime.h>
#include <hip/hip_bf16.h>

// ---------------------------------------------------------------------------
// Types for CDNA5 WMMA (gfx1250, wave32)
// ---------------------------------------------------------------------------
typedef __attribute__((ext_vector_type(8)))  __bf16 bf16x8;
typedef __attribute__((ext_vector_type(16))) __bf16 bf16x16;
typedef __attribute__((ext_vector_type(8)))  float  f32x8;
typedef int v4i __attribute__((vector_size(16)));

#define SHUF16(lo, hi) __builtin_shufflevector(lo, hi, 0,1,2,3,4,5,6,7,8,9,10,11,12,13,14,15)

static constexpr int B_BATCH = 4096;
static constexpr int T_SEQ   = 80;
static constexpr int EMBED   = 100;
static constexpr int KPAD0   = 128;        // padded layer-0 input dim
static constexpr int UNITS   = 512;
static constexpr int NTOT    = 4 * UNITS;  // 2048

// Async global->LDS copies (CDNA5); falls back to register staging if absent.
// Builtin signature (from clang diagnostics): (v4i AS1*, v4i AS3*, Imm offset, Imm cpol)
#if defined(__AMDGCN__) && __has_builtin(__builtin_amdgcn_global_load_async_to_lds_b128)
#define USE_ASYNC_LDS 1
#define GAS1(p) ((__attribute__((address_space(1))) v4i*)(p))
#define LAS3(p) ((__attribute__((address_space(3))) v4i*)(p))
#else
#define USE_ASYNC_LDS 0
#endif

// ---- bf16 <-> f32 via bits (round-to-nearest-even)
__device__ __forceinline__ unsigned short f2bf_bits(float f) {
    union { float f; unsigned u; } v; v.f = f;
    unsigned r = v.u + 0x7FFFu + ((v.u >> 16) & 1u);
    return (unsigned short)(r >> 16);
}
__device__ __forceinline__ float bfbits2f(unsigned short h) {
    union { float f; unsigned u; } v; v.u = ((unsigned)h) << 16;
    return v.f;
}

// ---- deterministic dropout mask hash, keep=0.8 -> {1.25, 0}
__device__ __forceinline__ float gate_mask(unsigned seed, int g, int b, int k) {
    unsigned x = seed ^ ((unsigned)g * 0x85EBCA6Bu) ^ ((unsigned)b * 0xC2B2AE35u)
                      ^ ((unsigned)k * 0x27D4EB2Fu);
    x ^= x >> 16; x *= 0x7FEB352Du; x ^= x >> 15; x *= 0x846CA68Bu; x ^= x >> 16;
    return (x < 3435973837u) ? 1.25f : 0.0f;
}

__device__ __forceinline__ float sigmoidf(float x) {
    return 1.0f / (1.0f + __expf(-x));
}

// ---------------------------------------------------------------------------
// Prep kernels
// ---------------------------------------------------------------------------
// transpose + convert weights f32 [K][2048] -> bf16 bits [2048][Kpad]
__global__ void conv_transpose_kernel(const float* __restrict__ in,
                                      unsigned short* __restrict__ out,
                                      int K, int Kpad) {
    int n = blockIdx.x;
    for (int k = threadIdx.x; k < Kpad; k += blockDim.x) {
        float v = (k < K) ? in[(long)k * NTOT + n] : 0.0f;
        out[(long)n * Kpad + k] = f2bf_bits(v);
    }
}

// embedding gather for timestep t -> 4 masked bf16 copies xm0 [4][B][128]
__global__ void embed_step_kernel(const float* __restrict__ embed,
                                  const int* __restrict__ seq, int t,
                                  unsigned seed,
                                  unsigned short* __restrict__ xm) {
    int idx = blockIdx.x * 256 + threadIdx.x;   // 4*B*128
    int g = idx >> 19;                          // B*128 = 2^19
    int r = idx & ((1 << 19) - 1);
    int b = r >> 7;
    int k = r & 127;
    int row = seq[(long)b * T_SEQ + t];
    float v = (k < EMBED) ? embed[(long)row * EMBED + k] : 0.0f;
    xm[idx] = f2bf_bits(v * gate_mask(seed, g, b, k));
}

__global__ void zero_u32_kernel(unsigned* __restrict__ p, int n) {
    int i = blockIdx.x * blockDim.x + threadIdx.x;
    if (i < n) p[i] = 0u;
}

// ---------------------------------------------------------------------------
// WMMA GEMM step:  z[b, n] = bias[n] + xmasked_g @ W + h @ U
// Block tile 256(M) x 128(N), 8 waves as 4x2, wave tile 64x64
// (4x4 wmma 16x16x32 bf16, B-fragments streamed).
// Double-buffered LDS; async global->LDS pipeline when available.
// ---------------------------------------------------------------------------
#define LDS_STRIDE 40   // padded K stride (elements): conflict-free, rows 80B (16B-mult)
static constexpr int BM = 256;
static constexpr int BN = 128;

__global__ __launch_bounds__(256)
void lstm_gemm_kernel(const unsigned short* __restrict__ Wt, int K1,
                      const unsigned short* __restrict__ Ut,
                      const float* __restrict__ bias,
                      const unsigned short* __restrict__ xm,    // bf16 [4][B][K1] (pre-masked)
                      const unsigned short* __restrict__ hrec,  // bf16 [B][512]
                      float* __restrict__ z) {
    const int tid   = threadIdx.x;
    const int nTile = blockIdx.x;          // 0..15
    const int mTile = blockIdx.y;          // 0..15
    const int gate  = nTile >> 2;
    const int rowBase = mTile * BM;
    const int colBase = nTile * BN;

    __shared__ unsigned short sA[2][BM * LDS_STRIDE];   // 40 KB
    __shared__ unsigned short sB[2][BN * LDS_STRIDE];   // 20 KB

    f32x8 acc[4][4] = {};

    const int lane  = tid & 31;
    const int wave  = tid >> 5;
    const int waveM = wave >> 1;           // 0..3
    const int waveN = wave & 1;            // 0..1
    const int l16   = lane & 15;
    const int hi    = lane >> 4;

    const int nch1 = K1 / 32;              // masked-input chunks
    const int nch  = nch1 + UNITS / 32;    // + recurrent chunks

    const unsigned short* xg = xm + (size_t)gate * B_BATCH * K1;

    // Loader mapping: A -> thread = row (256 rows x 32 K, 64B each)
    //                 B -> row = tid>>1, 16-elem half = tid&1
    const int rowB = tid >> 1;
    const int segB = tid & 1;

    auto srcA = [&](int c) -> const unsigned short* {
        return (c < nch1)
            ? xg   + (size_t)(rowBase + tid) * K1    + c * 32
            : hrec + (size_t)(rowBase + tid) * UNITS + (c - nch1) * 32;
    };
    auto srcB = [&](int c) -> const unsigned short* {
        return (c < nch1)
            ? Wt + (size_t)(colBase + rowB) * K1    + c * 32 + segB * 16
            : Ut + (size_t)(colBase + rowB) * UNITS + (c - nch1) * 32 + segB * 16;
    };

    auto compute = [&](int buf) {
        const unsigned short* A  = sA[buf];
        const unsigned short* Bt = sB[buf];
        bf16x16 afr[4];
        const int kA = hi * 8;             // A: lanes>=16 hold K 8-15 / 24-31
        #pragma unroll
        for (int i = 0; i < 4; ++i) {
            int row = waveM * 64 + i * 16 + l16;
            bf16x8 lo = *(const bf16x8*)&A[row * LDS_STRIDE + kA];
            bf16x8 h8 = *(const bf16x8*)&A[row * LDS_STRIDE + kA + 16];
            afr[i] = SHUF16(lo, h8);
        }
        const int kB = hi * 8;             // B: lanes>=16 hold K 8-15 / 24-31
        #pragma unroll
        for (int j = 0; j < 4; ++j) {
            int col = waveN * 64 + j * 16 + l16;
            bf16x8 lo = *(const bf16x8*)&Bt[col * LDS_STRIDE + kB];
            bf16x8 h8 = *(const bf16x8*)&Bt[col * LDS_STRIDE + kB + 16];
            bf16x16 bfr = SHUF16(lo, h8);
            #pragma unroll
            for (int i = 0; i < 4; ++i)
                acc[i][j] = __builtin_amdgcn_wmma_f32_16x16x32_bf16(
                    false, afr[i], false, bfr, (short)0, acc[i][j], false, false);
        }
    };

#if USE_ASYNC_LDS
    auto issue = [&](int c, int buf) {
        const unsigned short* pa = srcA(c);
        unsigned short* la = &sA[buf][tid * LDS_STRIDE];
        __builtin_amdgcn_global_load_async_to_lds_b128(GAS1(pa),      LAS3(la),      0, 0);
        __builtin_amdgcn_global_load_async_to_lds_b128(GAS1(pa + 8),  LAS3(la + 8),  0, 0);
        __builtin_amdgcn_global_load_async_to_lds_b128(GAS1(pa + 16), LAS3(la + 16), 0, 0);
        __builtin_amdgcn_global_load_async_to_lds_b128(GAS1(pa + 24), LAS3(la + 24), 0, 0);
        const unsigned short* pb = srcB(c);
        unsigned short* lb = &sB[buf][rowB * LDS_STRIDE + segB * 16];
        __builtin_amdgcn_global_load_async_to_lds_b128(GAS1(pb),     LAS3(lb),     0, 0);
        __builtin_amdgcn_global_load_async_to_lds_b128(GAS1(pb + 8), LAS3(lb + 8), 0, 0);
    };

    issue(0, 0);
    int buf = 0;
    for (int c = 0; c < nch; ++c) {
        asm volatile("s_wait_asynccnt 0" ::: "memory");   // our LDS writes landed
        __syncthreads();                                  // everyone's landed
        if (c + 1 < nch) issue(c + 1, buf ^ 1);
        compute(buf);
        buf ^= 1;
    }
#else
    uint4 ra[4], rb[2];
    auto load_regs = [&](int c) {
        const uint4* pa = (const uint4*)srcA(c);
        ra[0] = pa[0]; ra[1] = pa[1]; ra[2] = pa[2]; ra[3] = pa[3];
        const uint4* pb = (const uint4*)srcB(c);
        rb[0] = pb[0]; rb[1] = pb[1];
    };
    auto store_lds = [&](int buf) {
        uint4* pA = (uint4*)&sA[buf][tid * LDS_STRIDE];
        pA[0] = ra[0]; pA[1] = ra[1]; pA[2] = ra[2]; pA[3] = ra[3];
        uint4* pB = (uint4*)&sB[buf][rowB * LDS_STRIDE + segB * 16];
        pB[0] = rb[0]; pB[1] = rb[1];
    };

    load_regs(0);
    int buf = 0;
    store_lds(buf);
    for (int c = 0; c < nch; ++c) {
        __syncthreads();
        const bool more = (c + 1 < nch);
        if (more) load_regs(c + 1);
        compute(buf);
        if (more) { buf ^= 1; store_lds(buf); }
    }
#endif

    // ---- epilogue: z = acc + bias ----
    // C/D layout: VGPR v, lane<16 -> M=v,N=lane ; lane>=16 -> M=v+8,N=lane-16
    #pragma unroll
    for (int j = 0; j < 4; ++j) {
        const int col = colBase + waveN * 64 + j * 16 + l16;
        const float bc = bias[col];
        #pragma unroll
        for (int i = 0; i < 4; ++i) {
            const int rbase = rowBase + waveM * 64 + i * 16 + hi * 8;
            #pragma unroll
            for (int v = 0; v < 8; ++v)
                z[(size_t)(rbase + v) * NTOT + col] = acc[i][j][v] + bc;
        }
    }
}

// ---------------------------------------------------------------------------
// Elementwise LSTM gates; optionally also emits 4 per-gate masked bf16 copies
// of h (input-dropout for the next layer), mask fixed across time.
// ---------------------------------------------------------------------------
__global__ __launch_bounds__(256)
void lstm_gates_kernel(const float* __restrict__ z,
                       float* __restrict__ c,
                       unsigned short* __restrict__ hbf,
                       unsigned short* __restrict__ hm,  // nullable [4][B][512]
                       unsigned seed) {
    const int idx = blockIdx.x * 256 + threadIdx.x;   // B*UNITS
    const int b = idx >> 9;
    const int u = idx & 511;
    const float* zr = z + (size_t)b * NTOT;
    const float gi = sigmoidf(zr[u]);
    const float gf = sigmoidf(zr[512 + u]);
    const float gg = tanhf(zr[1024 + u]);
    const float go = sigmoidf(zr[1536 + u]);
    const float cn = gf * c[idx] + gi * gg;
    c[idx] = cn;
    const float hn = go * tanhf(cn);
    hbf[idx] = f2bf_bits(hn);
    if (hm) {
        #pragma unroll
        for (int g = 0; g < 4; ++g)
            hm[((size_t)g * B_BATCH + b) * UNITS + u] =
                f2bf_bits(hn * gate_mask(seed, g, b, u));
    }
}

// ---------------------------------------------------------------------------
// Output head: out[b] = sigmoid(h1[b,:] @ Wout + bout)
// ---------------------------------------------------------------------------
__global__ __launch_bounds__(256)
void lstm_out_kernel(const unsigned short* __restrict__ h1,
                     const float* __restrict__ Wout,
                     const float* __restrict__ bout,
                     float* __restrict__ out) {
    __shared__ float red[256];
    const int b = blockIdx.x;
    float s = 0.0f;
    for (int u = threadIdx.x; u < UNITS; u += 256)
        s += bfbits2f(h1[(size_t)b * UNITS + u]) * Wout[u];
    red[threadIdx.x] = s;
    __syncthreads();
    for (int w = 128; w > 0; w >>= 1) {
        if (threadIdx.x < w) red[threadIdx.x] += red[threadIdx.x + w];
        __syncthreads();
    }
    if (threadIdx.x == 0) out[b] = sigmoidf(red[0] + bout[0]);
}

// ---------------------------------------------------------------------------
// Host orchestration
// ---------------------------------------------------------------------------
extern "C" void kernel_launch(void* const* d_in, const int* in_sizes, int n_in,
                              void* d_out, int out_size, void* d_ws, size_t ws_size,
                              hipStream_t stream) {
    const int*   inputs = (const int*)  d_in[0];   // [B, T]
    const float* embed  = (const float*)d_in[1];   // [VOCAB, 100]
    const float* W0     = (const float*)d_in[2];   // [100, 2048]
    const float* U0     = (const float*)d_in[3];   // [512, 2048]
    const float* b0     = (const float*)d_in[4];   // [2048]
    const float* W1     = (const float*)d_in[5];   // [512, 2048]
    const float* U1     = (const float*)d_in[6];   // [512, 2048]
    const float* b1     = (const float*)d_in[7];   // [2048]
    const float* Wout   = (const float*)d_in[8];   // [512, 1]
    const float* bout   = (const float*)d_in[9];   // [1]
    float* out = (float*)d_out;                    // [B, 1]

    // ---- workspace carve ----
    char* ws = (char*)d_ws;
    unsigned short* Wt0 = (unsigned short*)ws; ws += (size_t)NTOT * KPAD0 * 2;        // 512 KB
    unsigned short* Ut0 = (unsigned short*)ws; ws += (size_t)NTOT * UNITS * 2;        // 2 MB
    unsigned short* Wt1 = (unsigned short*)ws; ws += (size_t)NTOT * UNITS * 2;        // 2 MB
    unsigned short* Ut1 = (unsigned short*)ws; ws += (size_t)NTOT * UNITS * 2;        // 2 MB
    unsigned short* xm0 = (unsigned short*)ws; ws += (size_t)4 * B_BATCH * KPAD0 * 2; // 4 MB
    unsigned short* hm0 = (unsigned short*)ws; ws += (size_t)4 * B_BATCH * UNITS * 2; // 16 MB
    float*          z   = (float*)ws;          ws += (size_t)B_BATCH * NTOT * 4;      // 32 MB
    float*          c0  = (float*)ws;          ws += (size_t)B_BATCH * UNITS * 4;     // 8 MB
    float*          c1  = (float*)ws;          ws += (size_t)B_BATCH * UNITS * 4;     // 8 MB
    unsigned short* h0  = (unsigned short*)ws; ws += (size_t)B_BATCH * UNITS * 2;     // 4 MB
    unsigned short* h1  = (unsigned short*)ws; ws += (size_t)B_BATCH * UNITS * 2;     // 4 MB

    const unsigned SEED0 = 0xA341316Cu;   // layer-0 input dropout (on embeddings)
    const unsigned SEED1 = 0xC8013EA4u;   // layer-1 input dropout (on h0)

    // ---- prep ----
    conv_transpose_kernel<<<NTOT, 128, 0, stream>>>(W0, Wt0, EMBED, KPAD0);
    conv_transpose_kernel<<<NTOT, 128, 0, stream>>>(U0, Ut0, UNITS, UNITS);
    conv_transpose_kernel<<<NTOT, 128, 0, stream>>>(W1, Wt1, UNITS, UNITS);
    conv_transpose_kernel<<<NTOT, 128, 0, stream>>>(U1, Ut1, UNITS, UNITS);

    // c0,c1 (f32) + h0,h1 (bf16) contiguous -> zero as u32 words
    const int zero_words = (B_BATCH * UNITS * 4 * 2 + B_BATCH * UNITS * 2 * 2) / 4;
    zero_u32_kernel<<<(zero_words + 255) / 256, 256, 0, stream>>>((unsigned*)c0, zero_words);

    // ---- time loop ----
    dim3 grid(NTOT / BN, B_BATCH / BM);    // (16, 16)
    const int gate_blocks = (B_BATCH * UNITS) / 256;
    for (int t = 0; t < T_SEQ; ++t) {
        embed_step_kernel<<<(4 * B_BATCH * KPAD0) / 256, 256, 0, stream>>>(
            embed, inputs, t, SEED0, xm0);

        lstm_gemm_kernel<<<grid, 256, 0, stream>>>(Wt0, KPAD0, Ut0, b0, xm0, h0, z);
        lstm_gates_kernel<<<gate_blocks, 256, 0, stream>>>(z, c0, h0, hm0, SEED1);

        lstm_gemm_kernel<<<grid, 256, 0, stream>>>(Wt1, UNITS, Ut1, b1, hm0, h1, z);
        lstm_gates_kernel<<<gate_blocks, 256, 0, stream>>>(z, c1, h1,
                                                           (unsigned short*)nullptr, 0u);
    }

    // ---- output head ----
    lstm_out_kernel<<<B_BATCH, 256, 0, stream>>>(h1, Wout, bout, out);
}
